// HGTModel_57019985822501
// MI455X (gfx1250) — matrix operand order repaired and checked by reading, not compiled, not run on hardware
//
#include <hip/hip_runtime.h>
#include <hip/hip_bf16.h>

// ---------------------------------------------------------------------------
// HGT forward for MI455X (gfx1250, wave32, WMMA).
// All dense linears (input proj, Q, fused per-relation K'/V', output proj) run
// through v_wmma_f32_16x16x32_bf16 with a copy-free ping-pong K pipeline.
// Edge softmax/aggregation is gather + L2 atomics (tables fit in 192MB L2).
// ---------------------------------------------------------------------------

typedef __bf16 v16bf __attribute__((ext_vector_type(16)));
typedef __bf16 v8bf  __attribute__((ext_vector_type(8)));
typedef float  v8f   __attribute__((ext_vector_type(8)));

__device__ inline __bf16 f2bf(float x) {
  unsigned int u = __float_as_uint(x);
  u = (u + 0x7FFFu + ((u >> 16) & 1u)) >> 16;   // round-to-nearest-even
  unsigned short s = (unsigned short)u;
  return __builtin_bit_cast(__bf16, s);
}

// order-preserving float <-> uint key for atomicMax-based segment max
__device__ inline unsigned int fkey(float x) {
  unsigned int u = __float_as_uint(x);
  return (u & 0x80000000u) ? ~u : (u | 0x80000000u);
}
__device__ inline float kdec(unsigned int k) {
  unsigned int u = (k & 0x80000000u) ? (k ^ 0x80000000u) : ~k;
  return __uint_as_float(u);   // key 0 -> NaN -> treated as "no edges" below
}

// A-fragment (16-bit 16x32): lane<16 row holds K {k0..k0+15};
// lanes>=16 hold K {k0+16..k0+31}. arow already includes half*16.
__device__ inline v16bf load_afrag(const __bf16* __restrict__ arow, int k0) {
  v8bf lo = *(const v8bf*)(arow + k0);
  v8bf hi = *(const v8bf*)(arow + k0 + 8);
  return __builtin_shufflevector(lo, hi, 0, 1, 2, 3, 4, 5, 6, 7,
                                         8, 9, 10, 11, 12, 13, 14, 15);
}

// ---------------------------------------------------------------------------
// elementwise helpers
// ---------------------------------------------------------------------------
__global__ __launch_bounds__(256) void f32_to_bf16_k(const float* __restrict__ x,
                                                     __bf16* __restrict__ y, size_t n) {
  size_t i = (size_t)blockIdx.x * 256 + threadIdx.x;
  if (i < n) y[i] = f2bf(x[i]);
}

__global__ __launch_bounds__(256) void zero_k(float* __restrict__ x, size_t n) {
  size_t i = (size_t)blockIdx.x * 256 + threadIdx.x;
  if (i < n) x[i] = 0.0f;
}

__global__ __launch_bounds__(256) void gelu_k(float* __restrict__ x, size_t n) {
  size_t i = (size_t)blockIdx.x * 256 + threadIdx.x;
  if (i < n) {
    float v = x[i];
    x[i] = 0.5f * v * (1.0f + erff(v * 0.7071067811865476f));  // exact gelu
  }
}

// W[K][N] f32 -> Wt[N][K] bf16 (transposed for WMMA B-fragment loads)
__global__ __launch_bounds__(256) void wtrans_k(const float* __restrict__ W,
                                                __bf16* __restrict__ Wt, int K, int N) {
  int i = blockIdx.x * 256 + threadIdx.x;
  if (i >= K * N) return;
  int n = i / K, k = i - n * K;
  Wt[i] = f2bf(W[(size_t)k * N + n]);
}

// Fuse the per-relation per-head DxD transform into the projection weight:
// Wt[n=h*32+f][k] = sum_d Wsrc[k][h*32+d] * a[h][d][f]   (output bf16, transposed)
__global__ __launch_bounds__(256) void wcomb_k(const float* __restrict__ Wsrc,
                                               const float* __restrict__ a,
                                               __bf16* __restrict__ Wt) {
  int i = blockIdx.x * 256 + threadIdx.x;
  if (i >= 256 * 256) return;
  int n = i >> 8, k = i & 255;
  int h = n >> 5, f = n & 31;
  const float* ah   = a + h * 1024;                 // a[h][d][f]
  const float* wrow = Wsrc + (size_t)k * 256 + h * 32;
  float s = 0.0f;
#pragma unroll 8
  for (int d = 0; d < 32; ++d) s += wrow[d] * ah[d * 32 + f];
  Wt[(size_t)n * 256 + k] = f2bf(s);
}

// combined bias: bt[n=h*32+f] = sum_d b[h*32+d] * a[h][d][f]
__global__ __launch_bounds__(256) void bcomb_k(const float* __restrict__ b,
                                               const float* __restrict__ a,
                                               float* __restrict__ bt) {
  int n = blockIdx.x * 256 + threadIdx.x;
  if (n >= 256) return;
  int h = n >> 5, f = n & 31;
  float s = 0.0f;
#pragma unroll 8
  for (int d = 0; d < 32; ++d) s += b[h * 32 + d] * a[h * 1024 + d * 32 + f];
  bt[n] = s;
}

// ---------------------------------------------------------------------------
// bf16 WMMA GEMM: C[M,N] = A[M,K] @ Bt[N,K]^T + bias, epilogue modes:
//   0: bias   1: bias+ReLU   2: bias then gated skip: g*o + (1-g)*xprev
// One wave computes a 32x64 tile: 2 A-frags x 4 B-frags = 8 WMMAs per K-step.
// K pipelined by 64 with two register fragment sets (no rotate copies): each
// WMMA group depends only on loads issued a half-iteration earlier, so waits
// are partial (10 newer loads outstanding). Requires N%64==0, K%64==0; M is
// row-guarded (loads clamp to M-1, stores guarded).
// ---------------------------------------------------------------------------
__global__ __launch_bounds__(256)
void gemm_bf16_k(const __bf16* __restrict__ A, const __bf16* __restrict__ Bt,
                 const float* __restrict__ bias, float* __restrict__ C,
                 int M, int N, int K, int mode,
                 const float* __restrict__ gate, const float* __restrict__ xprev) {
  const int lane = threadIdx.x & 31;
  const int wave = (blockIdx.x * 256 + threadIdx.x) >> 5;
  const int tiles_n = N >> 6;
  const int tiles_m = (M + 31) >> 5;
  const int tm = wave / tiles_n;
  if (tm >= tiles_m) return;
  const int tn = wave % tiles_n;
  const int mr = tm * 32;
  const int nc = tn * 64;
  const int half = lane >> 4;      // 0: lanes 0-15, 1: lanes 16-31
  const int l15  = lane & 15;

  // clamped A rows (duplicate reads for tail rows; never stored)
  int r0 = mr + l15;      if (r0 >= M) r0 = M - 1;
  int r1 = mr + 16 + l15; if (r1 >= M) r1 = M - 1;
  const __bf16* arow0 = A + (size_t)r0 * K + half * 16;
  const __bf16* arow1 = A + (size_t)r1 * K + half * 16;
  const __bf16* bptr[4];
#pragma unroll
  for (int t = 0; t < 4; ++t)
    bptr[t] = Bt + (size_t)(nc + 16 * t + l15) * K + half * 16; // lane = out col

  v8f acc[2][4];
#pragma unroll
  for (int f = 0; f < 2; ++f)
#pragma unroll
    for (int t = 0; t < 4; ++t) acc[f][t] = v8f{};

  // two fragment sets, ping-pong without register copies
  v16bf a0[2], a1[2], b[2][4];
  a0[0] = load_afrag(arow0, 0);
  a1[0] = load_afrag(arow1, 0);
#pragma unroll
  for (int t = 0; t < 4; ++t) b[0][t] = *(const v16bf*)(bptr[t]);

  for (int k0 = 0; k0 < K; k0 += 64) {
    const int k1 = k0 + 32;
    // issue set-1 loads (overlap with set-0 WMMAs below)
    __builtin_prefetch(arow0 + k0 + 128, 0, 2);
    __builtin_prefetch(arow1 + k0 + 128, 0, 2);
    a0[1] = load_afrag(arow0, k1);
    a1[1] = load_afrag(arow1, k1);
#pragma unroll
    for (int t = 0; t < 4; ++t) b[1][t] = *(const v16bf*)(bptr[t] + k1);
    // 8 WMMAs on set 0 (data loaded one half-iteration ago)
#pragma unroll
    for (int t = 0; t < 4; ++t)
      acc[0][t] = __builtin_amdgcn_wmma_f32_16x16x32_bf16(
          false, a0[0], false, b[0][t], (short)0, acc[0][t], false, false);
#pragma unroll
    for (int t = 0; t < 4; ++t)
      acc[1][t] = __builtin_amdgcn_wmma_f32_16x16x32_bf16(
          false, a1[0], false, b[0][t], (short)0, acc[1][t], false, false);
    // issue set-0 loads for k0+64 (clamped to 0 on last iteration: redundant,
    // branch-free, never consumed)
    const int k2 = (k0 + 64 < K) ? (k0 + 64) : 0;
    a0[0] = load_afrag(arow0, k2);
    a1[0] = load_afrag(arow1, k2);
#pragma unroll
    for (int t = 0; t < 4; ++t) b[0][t] = *(const v16bf*)(bptr[t] + k2);
    // 8 WMMAs on set 1
#pragma unroll
    for (int t = 0; t < 4; ++t)
      acc[0][t] = __builtin_amdgcn_wmma_f32_16x16x32_bf16(
          false, a0[1], false, b[1][t], (short)0, acc[0][t], false, false);
#pragma unroll
    for (int t = 0; t < 4; ++t)
      acc[1][t] = __builtin_amdgcn_wmma_f32_16x16x32_bf16(
          false, a1[1], false, b[1][t], (short)0, acc[1][t], false, false);
  }

  float g = 1.0f, gm = 0.0f;
  if (mode == 2) { g = 1.0f / (1.0f + __expf(-gate[0])); gm = 1.0f - g; }
#pragma unroll
  for (int f = 0; f < 2; ++f) {
    const int rb = mr + f * 16 + half * 8;  // C frag: VGPR j -> row rb+j
#pragma unroll
    for (int t = 0; t < 4; ++t) {
      const int col = nc + 16 * t + l15;
      const float bv = bias[col];
#pragma unroll
      for (int j = 0; j < 8; ++j) {
        const int row = rb + j;
        if (row < M) {
          size_t o = (size_t)row * N + col;
          float v = acc[f][t][j] + bv;
          if (mode == 1)      v = fmaxf(v, 0.0f);
          else if (mode == 2) v = g * v + gm * xprev[o];
          C[o] = v;
        }
      }
    }
  }
}

// ---------------------------------------------------------------------------
// Edge pass 1: per-edge per-head attention scores + segment max (atomic key).
// One wave per edge; lane = element-within-head, register j = head.
// ---------------------------------------------------------------------------
__global__ __launch_bounds__(256)
void edge_scores_k(const int* __restrict__ ei, int E,
                   const float* __restrict__ Q, const float* __restrict__ Kp,
                   const float* __restrict__ p, float scale,
                   float* __restrict__ sc, unsigned int* __restrict__ mkey) {
  const int wave = (blockIdx.x * 256 + threadIdx.x) >> 5;
  const int lane = threadIdx.x & 31;
  if (wave >= E) return;
  const int s = ei[wave];       // src row
  const int d = ei[E + wave];   // dst row
  const float* q = Q + (size_t)d * 256;
  const float* k = Kp + (size_t)s * 256;
  float pr[8];
#pragma unroll
  for (int j = 0; j < 8; ++j) { int c = lane + 32 * j; pr[j] = q[c] * k[c]; }
#pragma unroll
  for (int off = 16; off; off >>= 1)
#pragma unroll
    for (int j = 0; j < 8; ++j) pr[j] += __shfl_xor(pr[j], off, 32);
  if (lane == 0) {
#pragma unroll
    for (int j = 0; j < 8; ++j) {
      float v = pr[j] * p[j] * scale;
      sc[(size_t)wave * 8 + j] = v;
      atomicMax(&mkey[(size_t)d * 8 + j], fkey(v));
    }
  }
}

// Edge pass 2: e = exp(sc - max); segment sum. (max buffers complete: stream order)
__global__ __launch_bounds__(256)
void edge_exp_k(const int* __restrict__ ei, int E,
                const unsigned int* __restrict__ mkey,
                float* __restrict__ sc, float* __restrict__ ssum) {
  int i = blockIdx.x * 256 + threadIdx.x;
  if (i >= E * 8) return;
  int e = i >> 3, h = i & 7;
  int d = ei[E + e];
  float m = kdec(mkey[(size_t)d * 8 + h]);
  if (!isfinite(m)) m = 0.0f;        // matches jnp.where(isfinite(m), m, 0)
  float ev = __expf(sc[i] - m);
  sc[i] = ev;
  atomicAdd(&ssum[(size_t)d * 8 + h], ev);
}

// Edge pass 3: agg[dst] += alpha * V'[src]  (wave per edge, float atomics in L2)
__global__ __launch_bounds__(256)
void edge_agg_k(const int* __restrict__ ei, int E,
                const float* __restrict__ sc, const float* __restrict__ ssum,
                const float* __restrict__ Vp, float* __restrict__ agg) {
  const int wave = (blockIdx.x * 256 + threadIdx.x) >> 5;
  const int lane = threadIdx.x & 31;
  if (wave >= E) return;
  const int s = ei[wave];
  const int d = ei[E + wave];
  float a8 = 0.0f;
  if (lane < 8)
    a8 = sc[(size_t)wave * 8 + lane] / (ssum[(size_t)d * 8 + lane] + 1e-16f);
  const float* v = Vp + (size_t)s * 256;
  float* ag = agg + (size_t)d * 256;
#pragma unroll
  for (int j = 0; j < 8; ++j) {
    float alpha = __shfl(a8, j, 32);
    int c = lane + 32 * j;                 // head j, element lane
    atomicAdd(&ag[c], alpha * v[c]);
  }
}

// Final head: out[r][c] = h[r] . W[:,c] + b[c], W is (256,2)
__global__ __launch_bounds__(256)
void out_proj_k(const float* __restrict__ h, const float* __restrict__ W,
                const float* __restrict__ b, float* __restrict__ out, int M) {
  int i = blockIdx.x * 256 + threadIdx.x;
  if (i >= M * 2) return;
  int r = i >> 1, c = i & 1;
  const float* hr = h + (size_t)r * 256;
  float s = b[c];
#pragma unroll 8
  for (int k = 0; k < 256; ++k) s += hr[k] * W[k * 2 + c];
  out[i] = s;
}

// ---------------------------------------------------------------------------
// Host orchestration
// ---------------------------------------------------------------------------
extern "C" void kernel_launch(void* const* d_in, const int* in_sizes, int n_in,
                              void* d_out, int out_size, void* d_ws, size_t ws_size,
                              hipStream_t stream) {
  (void)in_sizes; (void)n_in; (void)out_size; (void)ws_size;
  const int Nn = 10000, Ni = 50000;
  const int E0 = 320000, E1 = 250000, E2 = 250000;
  const float inv_sqrt_d = 0.1767766952966369f;  // 1/sqrt(32)

  // Input order: x_news, x_interaction, ei_news_news, ei_int_news, ei_news_int,
  // then params leaves in JAX pytree order (nested dicts sorted by key):
  // per layer l (base 5+27l): kqv.interaction{Wk,Wq,Wv,bk,bq,bv},
  //   kqv.news{...}, out.interaction{W,b}, out.news{W,b},
  //   rel0{a,m,p}, rel1{a,m,p}, rel2{a,m,p}, skip.interaction, skip.news;
  // then lin.interaction{W,b}, lin.news{W,b}, out_lin{W,b}.
  auto F = [&](int i) { return (const float*)d_in[i]; };
  auto I = [&](int i) { return (const int*)d_in[i]; };

  char* base = (char*)d_ws;
  size_t off = 0;
  auto alloc = [&](size_t bytes) -> void* {
    void* p = base + off;
    off = (off + bytes + 255) & ~(size_t)255;
    return p;
  };

  float*  hA_n = (float*)alloc((size_t)Nn * 256 * 4);
  float*  hA_i = (float*)alloc((size_t)Ni * 256 * 4);
  float*  hB_n = (float*)alloc((size_t)Nn * 256 * 4);
  float*  hB_i = (float*)alloc((size_t)Ni * 256 * 4);
  __bf16* xb_n = (__bf16*)alloc((size_t)Nn * 768 * 2);
  __bf16* xb_i = (__bf16*)alloc((size_t)Ni * 384 * 2);
  __bf16* hb_n = (__bf16*)alloc((size_t)Nn * 256 * 2);
  __bf16* hb_i = (__bf16*)alloc((size_t)Ni * 256 * 2);
  __bf16* WtLn = (__bf16*)alloc((size_t)256 * 768 * 2);
  __bf16* WtLi = (__bf16*)alloc((size_t)256 * 384 * 2);
  __bf16* WtQn = (__bf16*)alloc(65536 * 2);
  __bf16* WtQi = (__bf16*)alloc(65536 * 2);
  __bf16* WtK0 = (__bf16*)alloc(65536 * 2);
  __bf16* WtK1 = (__bf16*)alloc(65536 * 2);
  __bf16* WtK2 = (__bf16*)alloc(65536 * 2);
  __bf16* WtV0 = (__bf16*)alloc(65536 * 2);
  __bf16* WtV1 = (__bf16*)alloc(65536 * 2);
  __bf16* WtV2 = (__bf16*)alloc(65536 * 2);
  __bf16* WtOn = (__bf16*)alloc(65536 * 2);
  __bf16* WtOi = (__bf16*)alloc(65536 * 2);
  float* btK0 = (float*)alloc(256 * 4);
  float* btK1 = (float*)alloc(256 * 4);
  float* btK2 = (float*)alloc(256 * 4);
  float* btV0 = (float*)alloc(256 * 4);
  float* btV1 = (float*)alloc(256 * 4);
  float* btV2 = (float*)alloc(256 * 4);
  float* Qn   = (float*)alloc((size_t)Nn * 256 * 4);
  float* Qi   = (float*)alloc((size_t)Ni * 256 * 4);
  float* Kp0  = (float*)alloc((size_t)Nn * 256 * 4);
  float* Kp1  = (float*)alloc((size_t)Ni * 256 * 4);
  float* Kp2  = (float*)alloc((size_t)Nn * 256 * 4);
  float* Vp0  = (float*)alloc((size_t)Nn * 256 * 4);
  float* Vp1  = (float*)alloc((size_t)Ni * 256 * 4);
  float* Vp2  = (float*)alloc((size_t)Nn * 256 * 4);
  float* aggn = (float*)alloc((size_t)Nn * 256 * 4);
  float* aggi = (float*)alloc((size_t)Ni * 256 * 4);
  float* scn  = (float*)alloc((size_t)(E0 + E1) * 8 * 4);
  float* sci  = (float*)alloc((size_t)E2 * 8 * 4);
  unsigned int* mkn = (unsigned int*)alloc((size_t)Nn * 8 * 4);
  unsigned int* mki = (unsigned int*)alloc((size_t)Ni * 8 * 4);
  float* smn = (float*)alloc((size_t)Nn * 8 * 4);
  float* smi = (float*)alloc((size_t)Ni * 8 * 4);

  auto conv = [&](const float* x, __bf16* y, size_t n) {
    f32_to_bf16_k<<<dim3((unsigned)((n + 255) / 256)), 256, 0, stream>>>(x, y, n);
  };
  auto wt = [&](const float* W, __bf16* Wt, int K, int N) {
    wtrans_k<<<dim3((unsigned)(((size_t)K * N + 255) / 256)), 256, 0, stream>>>(W, Wt, K, N);
  };
  auto gemm = [&](const __bf16* A, const __bf16* Bt, const float* bias, float* C,
                  int M, int N, int K, int mode, const float* gate, const float* xp) {
    int waves = ((M + 31) / 32) * (N / 64);
    gemm_bf16_k<<<dim3((unsigned)((waves + 7) / 8)), 256, 0, stream>>>(
        A, Bt, bias, C, M, N, K, mode, gate, xp);
  };
  auto zero = [&](void* p, size_t n) {
    zero_k<<<dim3((unsigned)((n + 255) / 256)), 256, 0, stream>>>((float*)p, n);
  };

  // ---- input projections: h = relu(x @ W + b) -----------------------------
  conv(F(0), xb_n, (size_t)Nn * 768);
  conv(F(1), xb_i, (size_t)Ni * 384);
  wt(F(61), WtLn, 768, 256);   // lin.news.W
  wt(F(59), WtLi, 384, 256);   // lin.interaction.W
  gemm(xb_n, WtLn, F(62), hA_n, Nn, 256, 768, 1, nullptr, nullptr);
  gemm(xb_i, WtLi, F(60), hA_i, Ni, 256, 384, 1, nullptr, nullptr);

  float* hn = hA_n; float* hin = hA_i; float* hnN = hB_n; float* hiN = hB_i;

  for (int l = 0; l < 2; ++l) {
    const int LB = 5 + 27 * l;
    const float *Wk_i = F(LB + 0), *Wq_i = F(LB + 1), *Wv_i = F(LB + 2);
    const float *bk_i = F(LB + 3), *bq_i = F(LB + 4), *bv_i = F(LB + 5);
    const float *Wk_n = F(LB + 6), *Wq_n = F(LB + 7), *Wv_n = F(LB + 8);
    const float *bk_n = F(LB + 9), *bq_n = F(LB + 10), *bv_n = F(LB + 11);
    const float *Wo_i = F(LB + 12), *bo_i = F(LB + 13);
    const float *Wo_n = F(LB + 14), *bo_n = F(LB + 15);
    const float *a0 = F(LB + 16), *m0 = F(LB + 17), *p0 = F(LB + 18);
    const float *a1 = F(LB + 19), *m1 = F(LB + 20), *p1 = F(LB + 21);
    const float *a2 = F(LB + 22), *m2 = F(LB + 23), *p2 = F(LB + 24);
    const float *sk_i = F(LB + 25), *sk_n = F(LB + 26);

    conv(hn, hb_n, (size_t)Nn * 256);
    conv(hin, hb_i, (size_t)Ni * 256);
    wt(Wq_n, WtQn, 256, 256);
    wt(Wq_i, WtQi, 256, 256);
    wt(Wo_n, WtOn, 256, 256);
    wt(Wo_i, WtOi, 256, 256);
    // fused relation weights (src types: rel0=news, rel1=interaction, rel2=news)
    wcomb_k<<<256, 256, 0, stream>>>(Wk_n, a0, WtK0);
    wcomb_k<<<256, 256, 0, stream>>>(Wk_i, a1, WtK1);
    wcomb_k<<<256, 256, 0, stream>>>(Wk_n, a2, WtK2);
    wcomb_k<<<256, 256, 0, stream>>>(Wv_n, m0, WtV0);
    wcomb_k<<<256, 256, 0, stream>>>(Wv_i, m1, WtV1);
    wcomb_k<<<256, 256, 0, stream>>>(Wv_n, m2, WtV2);
    bcomb_k<<<1, 256, 0, stream>>>(bk_n, a0, btK0);
    bcomb_k<<<1, 256, 0, stream>>>(bk_i, a1, btK1);
    bcomb_k<<<1, 256, 0, stream>>>(bk_n, a2, btK2);
    bcomb_k<<<1, 256, 0, stream>>>(bv_n, m0, btV0);
    bcomb_k<<<1, 256, 0, stream>>>(bv_i, m1, btV1);
    bcomb_k<<<1, 256, 0, stream>>>(bv_n, m2, btV2);

    // node-level projections (all WMMA)
    gemm(hb_n, WtQn, bq_n, Qn, Nn, 256, 256, 0, nullptr, nullptr);
    gemm(hb_i, WtQi, bq_i, Qi, Ni, 256, 256, 0, nullptr, nullptr);
    gemm(hb_n, WtK0, btK0, Kp0, Nn, 256, 256, 0, nullptr, nullptr);
    gemm(hb_i, WtK1, btK1, Kp1, Ni, 256, 256, 0, nullptr, nullptr);
    gemm(hb_n, WtK2, btK2, Kp2, Nn, 256, 256, 0, nullptr, nullptr);
    gemm(hb_n, WtV0, btV0, Vp0, Nn, 256, 256, 0, nullptr, nullptr);
    gemm(hb_i, WtV1, btV1, Vp1, Ni, 256, 256, 0, nullptr, nullptr);
    gemm(hb_n, WtV2, btV2, Vp2, Nn, 256, 256, 0, nullptr, nullptr);

    // segment-softmax state
    zero(mkn, (size_t)Nn * 8);
    zero(mki, (size_t)Ni * 8);
    zero(smn, (size_t)Nn * 8);
    zero(smi, (size_t)Ni * 8);
    zero(aggn, (size_t)Nn * 256);
    zero(aggi, (size_t)Ni * 256);

    // pass 1: scores + max (news dst gets rel0+rel1 concatenated)
    edge_scores_k<<<(E0 + 7) / 8, 256, 0, stream>>>(I(2), E0, Qn, Kp0, p0, inv_sqrt_d, scn, mkn);
    edge_scores_k<<<(E1 + 7) / 8, 256, 0, stream>>>(I(3), E1, Qn, Kp1, p1, inv_sqrt_d,
                                                    scn + (size_t)E0 * 8, mkn);
    edge_scores_k<<<(E2 + 7) / 8, 256, 0, stream>>>(I(4), E2, Qi, Kp2, p2, inv_sqrt_d, sci, mki);
    // pass 2: exp + sum
    edge_exp_k<<<(E0 * 8 + 255) / 256, 256, 0, stream>>>(I(2), E0, mkn, scn, smn);
    edge_exp_k<<<(E1 * 8 + 255) / 256, 256, 0, stream>>>(I(3), E1, mkn, scn + (size_t)E0 * 8, smn);
    edge_exp_k<<<(E2 * 8 + 255) / 256, 256, 0, stream>>>(I(4), E2, mki, sci, smi);
    // pass 3: weighted aggregation
    edge_agg_k<<<(E0 + 7) / 8, 256, 0, stream>>>(I(2), E0, scn, smn, Vp0, aggn);
    edge_agg_k<<<(E1 + 7) / 8, 256, 0, stream>>>(I(3), E1, scn + (size_t)E0 * 8, smn, Vp1, aggn);
    edge_agg_k<<<(E2 + 7) / 8, 256, 0, stream>>>(I(4), E2, sci, smi, Vp2, aggi);

    // gelu -> bf16 -> output proj with fused gated skip
    gelu_k<<<(unsigned)(((size_t)Nn * 256 + 255) / 256), 256, 0, stream>>>(aggn, (size_t)Nn * 256);
    gelu_k<<<(unsigned)(((size_t)Ni * 256 + 255) / 256), 256, 0, stream>>>(aggi, (size_t)Ni * 256);
    conv(aggn, hb_n, (size_t)Nn * 256);
    conv(aggi, hb_i, (size_t)Ni * 256);
    gemm(hb_n, WtOn, bo_n, hnN, Nn, 256, 256, 2, sk_n, hn);
    gemm(hb_i, WtOi, bo_i, hiN, Ni, 256, 256, 2, sk_i, hin);

    float* t;
    t = hn;  hn  = hnN; hnN = t;
    t = hin; hin = hiN; hiN = t;
  }

  // final head: [10000,256] @ [256,2] + b
  out_proj_k<<<(Nn * 2 + 255) / 256, 256, 0, stream>>>(hn, F(63), F(64), (float*)d_out, Nn);
}
